// MultiHeadAttention_4629974745245
// MI455X (gfx1250) — compile-verified, matrix operand order checked by reading, not run on hardware
//
#include <hip/hip_runtime.h>

typedef __attribute__((ext_vector_type(16))) __bf16 bf16x16;
typedef __attribute__((ext_vector_type(8)))  __bf16 bf16x8;
typedef __attribute__((ext_vector_type(8)))  float  f32x8;
typedef __attribute__((ext_vector_type(4)))  float  f32x4;

#define DEV static __device__ __forceinline__

#if __has_builtin(__builtin_amdgcn_sched_group_barrier)
#define SCHED_GROUP(mask, size) __builtin_amdgcn_sched_group_barrier(mask, size, 0)
#else
#define SCHED_GROUP(mask, size)
#endif
// mask bits: 0x002 VALU, 0x008 MFMA/WMMA, 0x020 VMEM read, 0x100 DS read

static constexpr int Bb = 2;
static constexpr int Hh = 16;
static constexpr int Ss = 2048;
static constexpr int Dm = 1024;
static constexpr int Dk = 64;

static constexpr int KC  = 128;      // staged k-chunk (double buffered)
static constexpr int LDW = KC + 8;   // padded LDS row stride (136 elem = 272B, 16B-aligned, banks rotate)

DEV f32x8 zero8() {
  f32x8 z;
#pragma unroll
  for (int i = 0; i < 8; ++i) z[i] = 0.0f;
  return z;
}

DEV f32x8 wmma_bf16(bf16x16 a, bf16x16 b, f32x8 c) {
  return __builtin_amdgcn_wmma_f32_16x16x32_bf16(
      false, a, false, b, (short)0, c, false, false);
}

// A/B fragment from row-major bf16 (ISA 16-bit 16x32 layout):
// lane<16: elems 0..7 = K k0+0..7, 8..15 = K k0+16..23; lane>=16: +8 in K.
DEV bf16x16 frag_from_bf16(const __bf16* base, int row0, int ld, int k0, int lane) {
  int r  = lane & 15;
  int ko = (lane >> 4) << 3;
  const __bf16* p = base + (size_t)(row0 + r) * ld + k0 + ko;
  bf16x8 lo = *(const bf16x8*)p;
  bf16x8 hi = *(const bf16x8*)(p + 16);
  return __builtin_shufflevector(lo, hi, 0, 1, 2, 3, 4, 5, 6, 7,
                                 8, 9, 10, 11, 12, 13, 14, 15);
}

// Same pattern from an fp32 source (4x16B loads + cvt).
DEV bf16x16 frag_from_f32(const float* base, int row0, int ld, int k0, int lane) {
  int r  = lane & 15;
  int ko = (lane >> 4) << 3;
  const float* p = base + (size_t)(row0 + r) * ld + k0 + ko;
  f32x4 a0 = *(const f32x4*)p;
  f32x4 a1 = *(const f32x4*)(p + 4);
  f32x4 b0 = *(const f32x4*)(p + 16);
  f32x4 b1 = *(const f32x4*)(p + 20);
  bf16x16 o;
#pragma unroll
  for (int i = 0; i < 4; ++i) {
    o[i]      = (__bf16)a0[i];
    o[i + 4]  = (__bf16)a1[i];
    o[i + 8]  = (__bf16)b0[i];
    o[i + 12] = (__bf16)b1[i];
  }
  return o;
}

// B fragment from a staged LDS tile [64][LDW].
DEV bf16x16 frag_from_lds(const __bf16* lds, int row0, int k0, int lane) {
  int r  = lane & 15;
  int ko = (lane >> 4) << 3;
  const __bf16* p = lds + (row0 + r) * LDW + k0 + ko;
  bf16x8 lo = *(const bf16x8*)p;
  bf16x8 hi = *(const bf16x8*)(p + 16);
  return __builtin_shufflevector(lo, hi, 0, 1, 2, 3, 4, 5, 6, 7,
                                 8, 9, 10, 11, 12, 13, 14, 15);
}

// Issue async copies of one 64-row x KC-col bf16 W chunk into LDS (no wait).
// 64*16 = 1024 16B chunks / 256 threads = 4 issues per thread.
DEV void stage_issue(__bf16* dst, const __bf16* Wb, int n0, int kc, int tid) {
  unsigned lbase = (unsigned)(uintptr_t)dst;
  const __bf16* gbase = Wb + (size_t)n0 * Dm + kc;   // uniform per block
#pragma unroll
  for (int pass = 0; pass < 4; ++pass) {
    int c   = tid + pass * 256;       // 0..1023
    int row = c >> 4;                 // 0..63
    int col = (c & 15) << 3;          // element offset 0..120
    unsigned laddr = lbase + (unsigned)(row * LDW + col) * 2u;
    unsigned voff  = (unsigned)(row * Dm + col) * 2u;
    asm volatile("global_load_async_to_lds_b128 %0, %1, %2"
                 :: "v"(laddr), "v"(voff), "s"(gbase) : "memory");
  }
}

// ---------------------------------------------------------------------------
// fp32 -> bf16 elementwise (weights only; 8 elements per thread)
// ---------------------------------------------------------------------------
__global__ void cvt_kernel(const float* __restrict__ src, __bf16* __restrict__ dst) {
  int i = (blockIdx.x * blockDim.x + threadIdx.x) << 3;
  f32x4 a = *(const f32x4*)(src + i);
  f32x4 b = *(const f32x4*)(src + i + 4);
  bf16x8 o;
#pragma unroll
  for (int j = 0; j < 4; ++j) {
    o[j]     = (__bf16)a[j];
    o[j + 4] = (__bf16)b[j];
  }
  *(bf16x8*)(dst + i) = o;
}

// ---------------------------------------------------------------------------
// Projection: Y = X (f32 [4096,1024]) * Wb^T (bf16 [1024,1024]) + bias -> bf16
// Block: 8 waves, M-tile 128 (16 per wave), N-tile 64 shared.
// W stripe staged into double-buffered LDS with async copies overlapping WMMA.
// vmode 0: store [B,H,S,Dk] (Q,K).  vmode 1: store [B,H,Dk,S] (V transposed).
// ---------------------------------------------------------------------------
__global__ void proj_kernel(const float* __restrict__ X, const __bf16* __restrict__ Wb,
                            const float* __restrict__ bias, __bf16* __restrict__ out,
                            int vmode) {
  __shared__ __align__(16) __bf16 Wl[2][64 * LDW];
  int tid  = threadIdx.x;
  int lane = tid & 31;
  int wave = tid >> 5;
  int n0 = (blockIdx.x & 15) << 6;                  // 16 N-stripes of 64
  int m0 = (blockIdx.x >> 4) * 128 + wave * 16;     // 32 M-blocks of 128

  f32x8 acc[4];
#pragma unroll
  for (int nt = 0; nt < 4; ++nt) acc[nt] = zero8();

  stage_issue(Wl[0], Wb, n0, 0, tid);
  int cur = 0;
  for (int kc = 0; kc < Dm; kc += KC) {
    asm volatile("s_wait_asynccnt 0x0" ::: "memory");
    __syncthreads();                                 // buf[cur] visible to all waves
    if (kc + KC < Dm) stage_issue(Wl[cur ^ 1], Wb, n0, kc + KC, tid);
    const __bf16* Wcur = Wl[cur];
#pragma unroll
    for (int k0 = 0; k0 < KC; k0 += 32) {
      bf16x16 af = frag_from_f32(X, m0, Dm, kc + k0, lane);
      bf16x16 b0 = frag_from_lds(Wcur, 0,  k0, lane);
      bf16x16 b1 = frag_from_lds(Wcur, 16, k0, lane);
      bf16x16 b2 = frag_from_lds(Wcur, 32, k0, lane);
      bf16x16 b3 = frag_from_lds(Wcur, 48, k0, lane);
      acc[0] = wmma_bf16(af, b0, acc[0]);
      acc[1] = wmma_bf16(af, b1, acc[1]);
      acc[2] = wmma_bf16(af, b2, acc[2]);
      acc[3] = wmma_bf16(af, b3, acc[3]);
      // Pin per-step schedule: A VMEM early, all 8 DS reads batched before the
      // 4 WMMAs so waits become partial and WMMA overlaps in-flight DS loads.
      SCHED_GROUP(0x020, 4);   // 4 VMEM reads (A tile, f32)
      SCHED_GROUP(0x100, 8);   // 8 DS reads (4 B fragments)
      SCHED_GROUP(0x002, 8);   // cvt_pk VALU batch
      SCHED_GROUP(0x008, 4);   // 4 WMMA
    }
    __syncthreads();                                 // all reads of buf[cur] done
    cur ^= 1;
  }

  int nloc = lane & 15;
  int mh   = (lane >> 4) << 3;
#pragma unroll
  for (int nt = 0; nt < 4; ++nt) {
    int n = n0 + nt * 16 + nloc;
    int h = n >> 6, d = n & 63;
    float bv = bias[n];
#pragma unroll
    for (int j = 0; j < 8; ++j) {
      int mg = m0 + mh + j;
      int b = mg >> 11, s = mg & 2047;
      float v = acc[nt][j] + bv;
      size_t idx;
      if (vmode == 0) idx = ((size_t)(b * Hh + h) * Ss + s) * Dk + d;
      else            idx = ((size_t)(b * Hh + h) * Dk + d) * Ss + s;
      out[idx] = (__bf16)v;
    }
  }
}

// ---------------------------------------------------------------------------
// Flash attention: one wave owns 16 query rows; block = 8 waves = 128 rows.
// grid.x = (B*H) * (S/128) = 512.
// ---------------------------------------------------------------------------
__global__ void attn_kernel(const __bf16* __restrict__ Q, const __bf16* __restrict__ K,
                            const __bf16* __restrict__ Vt, __bf16* __restrict__ ctx) {
  __shared__ __align__(16) __bf16 lds_p[8 * 16 * 40];  // per-wave 16x32 P tile, stride 40

  int lane = threadIdx.x & 31;
  int wave = threadIdx.x >> 5;
  int bh   = blockIdx.x >> 4;
  int qblk = blockIdx.x & 15;
  int qb   = qblk * 128 + wave * 16;

  const __bf16* Qh = Q  + (size_t)bh * Ss * Dk;
  const __bf16* Kh = K  + (size_t)bh * Ss * Dk;
  const __bf16* Vh = Vt + (size_t)bh * Dk * Ss;     // [Dk, Ss]
  __bf16* lp = lds_p + wave * (16 * 40);

  bf16x16 qf0 = frag_from_bf16(Qh, qb, Dk, 0,  lane);
  bf16x16 qf1 = frag_from_bf16(Qh, qb, Dk, 32, lane);

  f32x8 o0 = zero8(), o1 = zero8(), o2 = zero8(), o3 = zero8();
  float rmax[8], rsum[8];
#pragma unroll
  for (int j = 0; j < 8; ++j) { rmax[j] = -3.0e38f; rsum[j] = 0.0f; }

  int nloc = lane & 15;
  int mh   = (lane >> 4) << 3;
  int r    = lane & 15;
  int ko   = (lane >> 4) << 3;
  const float scale = 0.125f;      // 1/sqrt(Dk)

  for (int kb = 0; kb <= qb + 15; kb += 32) {
    // ---- scores: batch the 4 K-fragment loads, then 4 WMMAs ---------------
    f32x8 sc0 = zero8(), sc1 = zero8();
    {
      bf16x16 kf0 = frag_from_bf16(Kh, kb,      Dk, 0,  lane);
      bf16x16 kf1 = frag_from_bf16(Kh, kb + 16, Dk, 0,  lane);
      bf16x16 kf2 = frag_from_bf16(Kh, kb,      Dk, 32, lane);
      bf16x16 kf3 = frag_from_bf16(Kh, kb + 16, Dk, 32, lane);
      sc0 = wmma_bf16(qf0, kf0, sc0);
      sc1 = wmma_bf16(qf0, kf1, sc1);
      sc0 = wmma_bf16(qf1, kf2, sc0);
      sc1 = wmma_bf16(qf1, kf3, sc1);
    }

    // ---- prefetch V fragments; latency hides behind softmax VALU ----------
    bf16x16 vf0 = frag_from_bf16(Vh, 0,  Ss, kb, lane);
    bf16x16 vf1 = frag_from_bf16(Vh, 16, Ss, kb, lane);
    bf16x16 vf2 = frag_from_bf16(Vh, 32, Ss, kb, lane);
    bf16x16 vf3 = frag_from_bf16(Vh, 48, Ss, kb, lane);

    // ---- causal mask + online softmax ------------------------------------
    int kg0 = kb + nloc;
    int kg1 = kb + 16 + nloc;
#pragma unroll
    for (int j = 0; j < 8; ++j) {
      int qg = qb + mh + j;
      float s0 = sc0[j] * scale; if (kg0 > qg) s0 = -1.0e30f;
      float s1 = sc1[j] * scale; if (kg1 > qg) s1 = -1.0e30f;
      float v = fmaxf(s0, s1);
      v = fmaxf(v, __shfl_xor(v, 1));
      v = fmaxf(v, __shfl_xor(v, 2));
      v = fmaxf(v, __shfl_xor(v, 4));
      v = fmaxf(v, __shfl_xor(v, 8));
      float nm   = fmaxf(rmax[j], v);
      float corr = __expf(rmax[j] - nm);
      rmax[j] = nm;
      float p0 = __expf(s0 - nm);
      float p1 = __expf(s1 - nm);
      float ps = p0 + p1;
      ps += __shfl_xor(ps, 1);
      ps += __shfl_xor(ps, 2);
      ps += __shfl_xor(ps, 4);
      ps += __shfl_xor(ps, 8);
      rsum[j] = rsum[j] * corr + ps;
      o0[j] *= corr; o1[j] *= corr; o2[j] *= corr; o3[j] *= corr;
      sc0[j] = p0; sc1[j] = p1;
    }

    // ---- C-layout P -> LDS -> A-layout fragment (wave-private region) -----
#pragma unroll
    for (int j = 0; j < 8; ++j) {
      lp[(mh + j) * 40 + nloc]      = (__bf16)sc0[j];
      lp[(mh + j) * 40 + 16 + nloc] = (__bf16)sc1[j];
    }
    asm volatile("s_wait_dscnt 0" ::: "memory");
    bf16x16 pf;
    {
      const __bf16* pp = lp + r * 40 + ko;
      bf16x8 lo = *(const bf16x8*)pp;
      bf16x8 hi = *(const bf16x8*)(pp + 16);
      pf = __builtin_shufflevector(lo, hi, 0, 1, 2, 3, 4, 5, 6, 7,
                                   8, 9, 10, 11, 12, 13, 14, 15);
    }

    // ---- ctx += P(16x32) * V(32x64) ---------------------------------------
    o0 = wmma_bf16(pf, vf0, o0);
    o1 = wmma_bf16(pf, vf1, o1);
    o2 = wmma_bf16(pf, vf2, o2);
    o3 = wmma_bf16(pf, vf3, o3);
  }

  int b = bh >> 4, h = bh & 15;
#pragma unroll
  for (int j = 0; j < 8; ++j) {
    float inv = 1.0f / rsum[j];
    size_t row = (size_t)(b * Ss + qb + mh + j) * Dm + h * Dk;
    ctx[row + 0  + nloc] = (__bf16)(o0[j] * inv);
    ctx[row + 16 + nloc] = (__bf16)(o1[j] * inv);
    ctx[row + 32 + nloc] = (__bf16)(o2[j] * inv);
    ctx[row + 48 + nloc] = (__bf16)(o3[j] * inv);
  }
}

// ---------------------------------------------------------------------------
// Output projection: out = ctx (bf16 [4096,1024]) * Wob^T (bf16) + bo -> fp32
// Same double-buffered async-staged B-operand structure as proj_kernel.
// ---------------------------------------------------------------------------
__global__ void out_proj_kernel(const __bf16* __restrict__ Cx, const __bf16* __restrict__ Wob,
                                const float* __restrict__ bo, float* __restrict__ out) {
  __shared__ __align__(16) __bf16 Wl[2][64 * LDW];
  int tid  = threadIdx.x;
  int lane = tid & 31;
  int wave = tid >> 5;
  int n0 = (blockIdx.x & 15) << 6;
  int m0 = (blockIdx.x >> 4) * 128 + wave * 16;

  f32x8 acc[4];
#pragma unroll
  for (int nt = 0; nt < 4; ++nt) acc[nt] = zero8();

  stage_issue(Wl[0], Wob, n0, 0, tid);
  int cur = 0;
  for (int kc = 0; kc < Dm; kc += KC) {
    asm volatile("s_wait_asynccnt 0x0" ::: "memory");
    __syncthreads();
    if (kc + KC < Dm) stage_issue(Wl[cur ^ 1], Wob, n0, kc + KC, tid);
    const __bf16* Wcur = Wl[cur];
#pragma unroll
    for (int k0 = 0; k0 < KC; k0 += 32) {
      bf16x16 af = frag_from_bf16(Cx, m0, Dm, kc + k0, lane);
      bf16x16 b0 = frag_from_lds(Wcur, 0,  k0, lane);
      bf16x16 b1 = frag_from_lds(Wcur, 16, k0, lane);
      bf16x16 b2 = frag_from_lds(Wcur, 32, k0, lane);
      bf16x16 b3 = frag_from_lds(Wcur, 48, k0, lane);
      acc[0] = wmma_bf16(af, b0, acc[0]);
      acc[1] = wmma_bf16(af, b1, acc[1]);
      acc[2] = wmma_bf16(af, b2, acc[2]);
      acc[3] = wmma_bf16(af, b3, acc[3]);
      SCHED_GROUP(0x020, 2);   // 2 VMEM reads (A tile, bf16)
      SCHED_GROUP(0x100, 8);   // 8 DS reads (4 B fragments)
      SCHED_GROUP(0x008, 4);   // 4 WMMA
    }
    __syncthreads();
    cur ^= 1;
  }

  int nloc = lane & 15;
  int mh   = (lane >> 4) << 3;
#pragma unroll
  for (int nt = 0; nt < 4; ++nt) {
    int n = n0 + nt * 16 + nloc;
    float bv = bo[n];
#pragma unroll
    for (int j = 0; j < 8; ++j) {
      size_t mg = (size_t)(m0 + mh + j);
      out[mg * Dm + n] = acc[nt][j] + bv;
    }
  }
}

// ---------------------------------------------------------------------------
extern "C" void kernel_launch(void* const* d_in, const int* in_sizes, int n_in,
                              void* d_out, int out_size, void* d_ws, size_t ws_size,
                              hipStream_t stream) {
  (void)in_sizes; (void)n_in; (void)out_size; (void)ws_size;
  const float* query = (const float*)d_in[0];
  const float* key   = (const float*)d_in[1];
  const float* value = (const float*)d_in[2];
  // d_in[3] is the causal tril mask; implemented analytically in attn_kernel.
  const float* Wq = (const float*)d_in[4];
  const float* bq = (const float*)d_in[5];
  const float* Wk = (const float*)d_in[6];
  const float* bk = (const float*)d_in[7];
  const float* Wv = (const float*)d_in[8];
  const float* bv = (const float*)d_in[9];
  const float* Wo = (const float*)d_in[10];
  const float* bo = (const float*)d_in[11];
  float* out = (float*)d_out;

  const size_t wel = (size_t)Dm * Dm;               // 1,048,576
  const size_t tel = (size_t)Bb * Hh * Ss * Dk;     // 4,194,304
  __bf16* Wqb = (__bf16*)d_ws;                      // bf16 weights
  __bf16* Wkb = Wqb + wel;
  __bf16* Wvb = Wkb + wel;
  __bf16* Wob = Wvb + wel;
  __bf16* Qp  = Wob + wel;                          // [B,H,S,Dk]
  __bf16* Kp  = Qp  + tel;                          // [B,H,S,Dk]
  __bf16* Vtp = Kp  + tel;                          // [B,H,Dk,S]
  __bf16* Cx  = Vtp + tel;                          // [B,S,H*Dk]

  dim3 blk(256);
  cvt_kernel<<<wel / (256 * 8), blk, 0, stream>>>(Wq, Wqb);
  cvt_kernel<<<wel / (256 * 8), blk, 0, stream>>>(Wk, Wkb);
  cvt_kernel<<<wel / (256 * 8), blk, 0, stream>>>(Wv, Wvb);
  cvt_kernel<<<wel / (256 * 8), blk, 0, stream>>>(Wo, Wob);

  proj_kernel<<<512, blk, 0, stream>>>(query, Wqb, bq, Qp, 0);
  proj_kernel<<<512, blk, 0, stream>>>(key,   Wkb, bk, Kp, 0);
  proj_kernel<<<512, blk, 0, stream>>>(value, Wvb, bv, Vtp, 1);
  attn_kernel<<<512, blk, 0, stream>>>(Qp, Kp, Vtp, Cx);
  out_proj_kernel<<<512, blk, 0, stream>>>(Cx, Wob, bo, out);
}
